// EuclidToHypAlign_38697655337255
// MI455X (gfx1250) — matrix operand order repaired and checked by reading, not compiled, not run on hardware
//
#include <hip/hip_runtime.h>
#include <math.h>

typedef __attribute__((ext_vector_type(16))) _Float16 v16h;
typedef __attribute__((ext_vector_type(8)))  float    v8f;

#define B_   8
#define N_   16
#define L_   64
#define NL_  1024      // N_*L_
#define A_   16
#define DE_  256
#define DH_  64
#define H_   4
#define SCALE_ (1.0f/128.0f)   // 1/(A * sqrt(hd)) = 1/(16*8)

// ---------------------------------------------------------------------------
// Kernel A: q = LN(curr_rho)@Wq.T ; w[b,h,a,e] = gk[e] * sum_d q[h*64+d]*Wk[h*64+d,e] * SCALE
// One block per (b,a), 256 threads (= d_e).
// ---------------------------------------------------------------------------
__global__ void __launch_bounds__(256) proj_w_kernel(
    const float* __restrict__ curr_rho,
    const float* __restrict__ Wq,
    const float* __restrict__ Wk,
    const float* __restrict__ gq, const float* __restrict__ bq,
    const float* __restrict__ gk,
    float* __restrict__ w)
{
    __shared__ float red[256];
    __shared__ float red2[256];
    __shared__ float lnb[256];
    __shared__ float qb[256];
    const int b = blockIdx.x >> 4;
    const int a = blockIdx.x & 15;
    const int t = threadIdx.x;

    float x = curr_rho[(b*A_ + a)*DE_ + t];
    red[t] = x; red2[t] = x*x;
    __syncthreads();
    for (int s = 128; s > 0; s >>= 1) {
        if (t < s) { red[t] += red[t+s]; red2[t] += red2[t+s]; }
        __syncthreads();
    }
    const float mean = red[0] * (1.0f/DE_);
    const float var  = red2[0] * (1.0f/DE_) - mean*mean;
    const float rstd = rsqrtf(var + 1e-5f);
    lnb[t] = (x - mean) * rstd * gq[t] + bq[t];
    __syncthreads();

    // q[t] = sum_e Wq[t,e] * ln[e]
    {
        float acc = 0.f;
        const float* wq = Wq + t*DE_;
        #pragma unroll 4
        for (int e = 0; e < DE_; ++e) acc += wq[e]*lnb[e];
        qb[t] = acc;
    }
    __syncthreads();

    // w[b,h,a,e] (1024 entries, 4 per thread)
    for (int i = t; i < H_*DE_; i += 256) {
        const int h = i >> 8, e = i & 255;
        float s = 0.f;
        const float* qh = qb + h*DH_;
        const float* wk = Wk + (h*DH_)*DE_ + e;
        #pragma unroll 4
        for (int d = 0; d < DH_; ++d) s += qh[d] * wk[d*DE_];
        w[((b*H_ + h)*A_ + a)*DE_ + e] = s * gk[e] * SCALE_;
    }
}

// ---------------------------------------------------------------------------
// Kernel B: fused LN + skinny GEMM via v_wmma_f32_16x16x32_f16, with
// double-buffered GLOBAL_LOAD_ASYNC_TO_LDS_B128 staging (ASYNCcnt pipeline).
// scores[b,h,nl] = sum_a sum_e (x-mu)*rstd * w[b,h,a,e]
// One wave per block; each block owns one 16-row (n,l) tile; grid = B * 64.
// ---------------------------------------------------------------------------

// Issue this lane's 32 x b128 async copies (one row-half = 128 floats = 512B)
// into LDS.  offset: must be an immediate, hence the literal macro chain.
#define A_ISSUE(OFF) \
    asm volatile("global_load_async_to_lds_b128 %0, %1, off offset:" #OFF \
                 :: "v"(ldsaddr), "v"(gaddr) : "memory")

__device__ __forceinline__ void async_copy_half_row(unsigned ldsaddr,
                                                    unsigned long long gaddr)
{
    A_ISSUE(0);   A_ISSUE(16);  A_ISSUE(32);  A_ISSUE(48);
    A_ISSUE(64);  A_ISSUE(80);  A_ISSUE(96);  A_ISSUE(112);
    A_ISSUE(128); A_ISSUE(144); A_ISSUE(160); A_ISSUE(176);
    A_ISSUE(192); A_ISSUE(208); A_ISSUE(224); A_ISSUE(240);
    A_ISSUE(256); A_ISSUE(272); A_ISSUE(288); A_ISSUE(304);
    A_ISSUE(320); A_ISSUE(336); A_ISSUE(352); A_ISSUE(368);
    A_ISSUE(384); A_ISSUE(400); A_ISSUE(416); A_ISSUE(432);
    A_ISSUE(448); A_ISSUE(464); A_ISSUE(480); A_ISSUE(496);
}

__global__ void __launch_bounds__(32) scores_wmma_kernel(
    const float* __restrict__ demo_rho,
    const float* __restrict__ w,
    float* __restrict__ scores)
{
    __shared__ float tile[2][16][260];   // double buffer; +4 pad kills bank conflicts
    const int lane  = threadIdx.x & 31;
    const int b     = blockIdx.x >> 6;
    const int tilei = blockIdx.x & 63;
    const int nl0   = tilei << 4;
    const int m     = lane & 15;         // A-matrix row / B-matrix column
    const int hsel  = lane >> 4;         // low/high half of wave

    // per-lane copy source: row (nl0+m), element half hsel*128, a-slice stride DE_
    const float* gbase = demo_rho + ((b*NL_ + nl0 + m)*A_)*DE_ + hsel*128;
    const unsigned lds0 = (unsigned)(size_t)&tile[0][m][hsel*128];
    const unsigned lds1 = (unsigned)(size_t)&tile[1][m][hsel*128];

    v8f c = {};

    // prime the pipeline with a = 0 into buffer 0
    async_copy_half_row(lds0, (unsigned long long)(size_t)gbase);

    for (int a = 0; a < A_; ++a) {
        const int cur = a & 1;
        // kick off next slice into the other buffer, then wait for current:
        // async loads retire in order, so <=32 outstanding means buffer 'cur'
        // is fully resident.
        if (a + 1 < A_) {
            async_copy_half_row(cur ? lds0 : lds1,
                                (unsigned long long)(size_t)(gbase + (a+1)*DE_));
            asm volatile("s_wait_asynccnt 32" ::: "memory");
        } else {
            asm volatile("s_wait_asynccnt 0" ::: "memory");
        }

        // ---- per-row mean / rstd (2 lanes per row, shfl_xor combine) ----
        float s = 0.f, s2 = 0.f;
        const int eb = hsel * 128;
        #pragma unroll 8
        for (int j = 0; j < 32; ++j) {
            float4 v = *(const float4*)&tile[cur][m][eb + (j << 2)];
            s  += v.x + v.y + v.z + v.w;
            s2 += v.x*v.x + v.y*v.y + v.z*v.z + v.w*v.w;
        }
        s  += __shfl_xor(s, 16, 32);
        s2 += __shfl_xor(s2, 16, 32);
        const float mean = s  * (1.0f/DE_);
        const float var  = s2 * (1.0f/DE_) - mean*mean;
        const float rstd = rsqrtf(var + 1e-5f);

        // ---- 8 WMMA steps over K=256 ----
        for (int kb = 0; kb < DE_; kb += 32) {
            const int off = hsel << 3;       // A: low lanes K 0..7/16..23, high 8..15/24..31
            float4 a0 = *(const float4*)&tile[cur][m][kb + off];
            float4 a1 = *(const float4*)&tile[cur][m][kb + off + 4];
            float4 a2 = *(const float4*)&tile[cur][m][kb + 16 + off];
            float4 a3 = *(const float4*)&tile[cur][m][kb + 16 + off + 4];
            v16h hA;
            hA[0]  = (_Float16)((a0.x - mean) * rstd);
            hA[1]  = (_Float16)((a0.y - mean) * rstd);
            hA[2]  = (_Float16)((a0.z - mean) * rstd);
            hA[3]  = (_Float16)((a0.w - mean) * rstd);
            hA[4]  = (_Float16)((a1.x - mean) * rstd);
            hA[5]  = (_Float16)((a1.y - mean) * rstd);
            hA[6]  = (_Float16)((a1.z - mean) * rstd);
            hA[7]  = (_Float16)((a1.w - mean) * rstd);
            hA[8]  = (_Float16)((a2.x - mean) * rstd);
            hA[9]  = (_Float16)((a2.y - mean) * rstd);
            hA[10] = (_Float16)((a2.z - mean) * rstd);
            hA[11] = (_Float16)((a2.w - mean) * rstd);
            hA[12] = (_Float16)((a3.x - mean) * rstd);
            hA[13] = (_Float16)((a3.y - mean) * rstd);
            hA[14] = (_Float16)((a3.z - mean) * rstd);
            hA[15] = (_Float16)((a3.w - mean) * rstd);

            v16h hB;
            if (m < H_) {                    // columns 0..3 = heads, rest zero-pad
                const int off2 = hsel << 3;  // B: low lanes K 0..7/16..23, high 8..15/24..31
                const float* wrow = w + ((b*H_ + m)*A_ + a)*DE_ + kb + off2;
                float4 b0 = *(const float4*)(wrow);
                float4 b1 = *(const float4*)(wrow + 4);
                float4 b2 = *(const float4*)(wrow + 16);
                float4 b3 = *(const float4*)(wrow + 20);
                hB[0]  = (_Float16)b0.x; hB[1]  = (_Float16)b0.y;
                hB[2]  = (_Float16)b0.z; hB[3]  = (_Float16)b0.w;
                hB[4]  = (_Float16)b1.x; hB[5]  = (_Float16)b1.y;
                hB[6]  = (_Float16)b1.z; hB[7]  = (_Float16)b1.w;
                hB[8]  = (_Float16)b2.x; hB[9]  = (_Float16)b2.y;
                hB[10] = (_Float16)b2.z; hB[11] = (_Float16)b2.w;
                hB[12] = (_Float16)b3.x; hB[13] = (_Float16)b3.y;
                hB[14] = (_Float16)b3.z; hB[15] = (_Float16)b3.w;
            } else {
                #pragma unroll
                for (int i = 0; i < 16; ++i) hB[i] = (_Float16)0.0f;
            }

            c = __builtin_amdgcn_wmma_f32_16x16x32_f16(
                    false, hA, false, hB, (short)0, c, false, false);
        }
    }

    // C/D layout: VGPR r -> M = r + 8*hsel, N = lane&15
    if (m < H_) {
        float* out = scores + (b*H_ + m)*NL_ + nl0 + (hsel << 3);
        #pragma unroll
        for (int r = 0; r < 8; ++r) out[r] = c[r];
    }
}

// ---------------------------------------------------------------------------
// Kernel C: masked softmax over N*L = 1024 per (b,h). (bk term is a constant
// shift per (b,h) -> dropped by softmax invariance.)
// ---------------------------------------------------------------------------
__global__ void __launch_bounds__(256) softmax_kernel(
    const float* __restrict__ scores,
    const int* __restrict__ demo_mask,
    float* __restrict__ attn)
{
    __shared__ float red[256];
    const int b = blockIdx.x >> 2, h = blockIdx.x & 3, t = threadIdx.x;
    const float* s  = scores + (b*H_ + h)*NL_;
    const int*   mk = demo_mask + b*NL_;

    float vals[4];
    float mx = -__builtin_inff();
    #pragma unroll
    for (int j = 0; j < 4; ++j) {
        const int i = t + j*256;
        const float v = (mk[i] == 1) ? s[i] : -__builtin_inff();
        vals[j] = v;
        mx = fmaxf(mx, v);
    }
    red[t] = mx; __syncthreads();
    for (int st = 128; st > 0; st >>= 1) {
        if (t < st) red[t] = fmaxf(red[t], red[t+st]);
        __syncthreads();
    }
    mx = red[0]; __syncthreads();

    float sum = 0.f;
    #pragma unroll
    for (int j = 0; j < 4; ++j) {
        float e = (vals[j] == -__builtin_inff()) ? 0.0f : __expf(vals[j] - mx);
        vals[j] = e; sum += e;
    }
    red[t] = sum; __syncthreads();
    for (int st = 128; st > 0; st >>= 1) {
        if (t < st) red[t] += red[t+st];
        __syncthreads();
    }
    const float inv = 1.0f / red[0];
    #pragma unroll
    for (int j = 0; j < 4; ++j)
        attn[(b*H_ + h)*NL_ + t + j*256] = vals[j] * inv;
}

// ---------------------------------------------------------------------------
// Kernel D: Poincare aggregation. One block per b, 256 threads = (h,d).
// ---------------------------------------------------------------------------
__device__ __forceinline__ float artanh_over_n(float nrm) {
    const float arg = fminf(nrm, 1.0f - 1e-5f);
    return 0.5f * logf((1.0f + arg) / (1.0f - arg)) / nrm;
}

__global__ void __launch_bounds__(256) hyp_agg_kernel(
    const float* __restrict__ demo_hyp,
    const float* __restrict__ attn,
    float* __restrict__ out)
{
    __shared__ float fbuf[NL_];
    __shared__ float mh[H_][DH_];
    __shared__ float zz[H_][DH_];
    __shared__ float scal[H_];
    __shared__ float scal2[H_];
    const int b = blockIdx.x, t = threadIdx.x;

    // phase 1: logmap0 scale factors per (n,l) row
    for (int nl = t; nl < NL_; nl += 256) {
        const float* x = demo_hyp + (b*NL_ + nl)*DH_;
        float s = 0.f;
        #pragma unroll
        for (int d = 0; d < DH_; d += 4) {
            float4 v = *(const float4*)(x + d);
            s += v.x*v.x + v.y*v.y + v.z*v.z + v.w*v.w;
        }
        const float nrm = fmaxf(sqrtf(s), 1e-6f);
        fbuf[nl] = artanh_over_n(nrm);
    }
    __syncthreads();

    // phase 2: m_h[h,d] = sum_nl attn * f * x
    const int h = t >> 6, d = t & 63;
    {
        const float* at = attn + (b*H_ + h)*NL_;
        float acc = 0.f;
        for (int nl = 0; nl < NL_; ++nl)
            acc += at[nl] * fbuf[nl] * demo_hyp[(b*NL_ + nl)*DH_ + d];
        mh[h][d] = acc;
    }
    __syncthreads();

    // y_h = expmap0(m_h)
    if (t < H_) {
        float s = 0.f;
        for (int i = 0; i < DH_; ++i) s += mh[t][i]*mh[t][i];
        const float nrm = fmaxf(sqrtf(s), 1e-6f);
        scal[t] = tanhf(nrm) / nrm;
    }
    __syncthreads();
    zz[h][d] = mh[h][d] * scal[h];
    __syncthreads();

    // z_h = logmap0(y_h)
    if (t < H_) {
        float s = 0.f;
        for (int i = 0; i < DH_; ++i) s += zz[t][i]*zz[t][i];
        const float nrm = fmaxf(sqrtf(s), 1e-6f);
        scal2[t] = artanh_over_n(nrm);
    }
    __syncthreads();

    // m = mean_h z_h  (reuse fbuf[0..63])
    if (t < DH_) {
        float mval = 0.f;
        #pragma unroll
        for (int hh = 0; hh < H_; ++hh) mval += zz[hh][t] * scal2[hh];
        fbuf[t] = mval * (1.0f/H_);
    }
    __syncthreads();

    // curr_hyp = clip(expmap0(m))
    if (t == 0) {
        float s = 0.f;
        for (int i = 0; i < DH_; ++i) s += fbuf[i]*fbuf[i];
        const float nrm = fmaxf(sqrtf(s), 1e-6f);
        const float f = tanhf(nrm) / nrm;
        float s2 = 0.f;
        for (int i = 0; i < DH_; ++i) { const float c2 = fbuf[i]*f; s2 += c2*c2; }
        const float nrm2 = fmaxf(sqrtf(s2), 1e-6f);
        scal[0] = f * fminf((1.0f - 1e-5f) / nrm2, 1.0f);
    }
    __syncthreads();
    if (t < DH_) out[b*DH_ + t] = fbuf[t] * scal[0];
}

// ---------------------------------------------------------------------------
extern "C" void kernel_launch(void* const* d_in, const int* in_sizes, int n_in,
                              void* d_out, int out_size, void* d_ws, size_t ws_size,
                              hipStream_t stream)
{
    (void)in_sizes; (void)n_in; (void)out_size; (void)ws_size;
    const float* curr_rho  = (const float*)d_in[0];
    const float* demo_rho  = (const float*)d_in[1];
    const float* demo_hyp  = (const float*)d_in[2];
    const int*   demo_mask = (const int*)  d_in[3];
    const float* Wq = (const float*)d_in[4];
    const float* Wk = (const float*)d_in[5];
    const float* gq = (const float*)d_in[6];
    const float* bq = (const float*)d_in[7];
    const float* gk = (const float*)d_in[8];
    // d_in[9] (bk) contributes a per-(b,h) constant to scores -> softmax-invariant, dropped.

    float* w      = (float*)d_ws;                       // B*H*A*DE = 131072 f32
    float* scores = w + (B_*H_*A_*DE_);                 // B*H*NL   = 32768 f32
    float* out_hyp  = (float*)d_out;                    // [B, DH]  = 512 f32
    float* out_attn = out_hyp + B_*DH_;                 // [B,H,N,L]= 32768 f32

    proj_w_kernel     <<<B_*A_, 256, 0, stream>>>(curr_rho, Wq, Wk, gq, bq, gk, w);
    scores_wmma_kernel<<<B_*64,  32, 0, stream>>>(demo_rho, w, scores);
    softmax_kernel    <<<B_*H_, 256, 0, stream>>>(scores, demo_mask, out_attn);
    hyp_agg_kernel    <<<B_,    256, 0, stream>>>(demo_hyp, out_attn, out_hyp);
}
